// GRUNet_22806276342353
// MI455X (gfx1250) — matrix operand order repaired and checked by reading, not compiled, not run on hardware
//
#include <hip/hip_runtime.h>

// ---------------------------------------------------------------------------
// 2-layer GRU (B=256, S=1200, D=32, H=128) + ReLU + Linear(128->1) for gfx1250.
// Batch-tiled (M=16) persistent recurrence kernels, one per layer.
// v_wmma_f32_16x16x32_f16 everywhere; recurrent weights live in VGPRs
// (loop-invariant B fragments), LDS state tiles are double-buffered so each
// timestep needs a single workgroup barrier.
// ---------------------------------------------------------------------------

typedef _Float16 half_t;
typedef __attribute__((ext_vector_type(16))) _Float16 v16h;
typedef __attribute__((ext_vector_type(8)))  _Float16 v8h;
typedef __attribute__((ext_vector_type(8)))  float    v8f;

#define BATCH 256
#define SEQ   1200
#define HID   128
#define DIN   32
#define G3    384   // 3*HID, gates stacked (r,z,n)
#define TPB   256   // 8 waves of 32

__device__ __forceinline__ v8f wmma_f16(v16h a, v16h b, v8f c) {
  // D = A(16x32 f16) * B(32x16 f16) + C(16x16 f32)
  return __builtin_amdgcn_wmma_f32_16x16x32_f16(false, a, false, b, (short)0, c,
                                                false, false);
}

// A-matrix fragment (16x32 f16) from row-major LDS tile [16][ldk].
// lanes 0-15: row m, K = kbase+{0..15}; lanes 16-31: K=kbase+{16..31}.
__device__ __forceinline__ v16h load_a_frag(const half_t* lds, int ldk,
                                            int lane, int kbase) {
  const int m   = lane & 15;
  const int sel = lane >> 4;
  const half_t* p = lds + m * ldk + kbase + sel * 16;
  v8h lo = *(const v8h*)p;
  v8h hi = *(const v8h*)(p + 8);
  return __builtin_shufflevector(lo, hi, 0, 1, 2, 3, 4, 5, 6, 7,
                                         8, 9, 10, 11, 12, 13, 14, 15);
}

// B-matrix fragment (32x16 f16), B(k,n) = W[n][k], from f16 LDS [N][ldk].
__device__ __forceinline__ v16h load_b_frag(const half_t* ldsW, int ldk,
                                            int lane, int ntile, int kbase) {
  const int n = (ntile << 4) + (lane & 15);
  const half_t* p = ldsW + n * ldk + kbase + ((lane >> 4) << 4);
  return *(const v16h*)p;                    // 32B -> 2x ds_load_b128
}

// Same fragment, loaded once at startup directly from f32 global memory.
__device__ __forceinline__ v16h load_b_frag_gmem(const float* __restrict__ W,
                                                 int ldk, int lane, int ntile,
                                                 int kbase) {
  const int n = (ntile << 4) + (lane & 15);
  const float* p = W + (size_t)n * ldk + kbase + ((lane >> 4) << 4);
  float4 f0 = ((const float4*)p)[0];
  float4 f1 = ((const float4*)p)[1];
  float4 f2 = ((const float4*)p)[2];
  float4 f3 = ((const float4*)p)[3];
  v16h r;
  r[0]  = (_Float16)f0.x; r[1]  = (_Float16)f0.y;
  r[2]  = (_Float16)f0.z; r[3]  = (_Float16)f0.w;
  r[4]  = (_Float16)f1.x; r[5]  = (_Float16)f1.y;
  r[6]  = (_Float16)f1.z; r[7]  = (_Float16)f1.w;
  r[8]  = (_Float16)f2.x; r[9]  = (_Float16)f2.y;
  r[10] = (_Float16)f2.z; r[11] = (_Float16)f2.w;
  r[12] = (_Float16)f3.x; r[13] = (_Float16)f3.y;
  r[14] = (_Float16)f3.z; r[15] = (_Float16)f3.w;
  return r;
}

__device__ __forceinline__ float fast_sigmoid(float x) {
  return 1.0f / (1.0f + __expf(-x));
}
__device__ __forceinline__ float fast_tanh(float x) {
  x = fminf(fmaxf(x, -15.0f), 15.0f);
  float e = __expf(2.0f * x);
  return (e - 1.0f) / (e + 1.0f);
}

// ---------------------------------------------------------------------------
// Layer 0: x[B,S,32] f32 -> h1[S,B,128] f16 workspace.
// All 15 weight B-fragments register-resident; inner loop = 10 ds_load_b128
// (A fragments) + 15 back-to-back WMMAs. Single barrier per step.
// ---------------------------------------------------------------------------
__global__ __launch_bounds__(TPB) void gru_layer0_kernel(
    const float* __restrict__ x,
    const float* __restrict__ Wih,   // [384,32]
    const float* __restrict__ Whh,   // [384,128]
    const float* __restrict__ bih,
    const float* __restrict__ bhh,
    half_t* __restrict__ h1out) {
  __shared__ half_t lds_h[2][16 * HID];   // ping-pong recurrent state
  __shared__ half_t lds_x[2][16 * DIN];   // ping-pong staged input

  const int t    = threadIdx.x;
  const int lane = t & 31;
  const int w    = t >> 5;
  const int b0   = blockIdx.x * 16;

  // Loop-invariant weight fragments -> registers (f32 global, convert to f16).
  v16h bxr = load_b_frag_gmem(Wih, DIN, lane, w,      0);
  v16h bxz = load_b_frag_gmem(Wih, DIN, lane, 8 + w,  0);
  v16h bxn = load_b_frag_gmem(Wih, DIN, lane, 16 + w, 0);
  v16h bhr[4], bhz[4], bhn[4];
#pragma unroll
  for (int k = 0; k < 4; ++k) {
    bhr[k] = load_b_frag_gmem(Whh, HID, lane, w,      k * 32);
    bhz[k] = load_b_frag_gmem(Whh, HID, lane, 8 + w,  k * 32);
    bhn[k] = load_b_frag_gmem(Whh, HID, lane, 16 + w, k * 32);
  }

  {  // h(-1) = 0 in buffer 0
    uint4 z; z.x = z.y = z.z = z.w = 0u;
    ((uint4*)&lds_h[0][0])[t] = z;        // 256*16B = 4096B
  }
  {  // stage x(0) into buffer 0
    const int m = t >> 4, c = t & 15;
    const float2 xv = *(const float2*)(x + ((size_t)(b0 + m) * SEQ) * DIN + 2 * c);
    lds_x[0][m * DIN + 2 * c]     = (half_t)xv.x;
    lds_x[0][m * DIN + 2 * c + 1] = (half_t)xv.y;
  }

  const int   j   = (w << 4) + (lane & 15);
  const float br  = bih[j] + bhh[j];
  const float bz  = bih[HID + j] + bhh[HID + j];
  const float bxnb = bih[2 * HID + j];
  const float bhnb = bhh[2 * HID + j];
  const int   mb  = (lane >> 4) << 3;

  float hprev[8];
#pragma unroll
  for (int v = 0; v < 8; ++v) hprev[v] = 0.0f;

  for (int s = 0; s < SEQ; ++s) {
    __syncthreads();                      // step-(s-1) writes visible
    const int p = s & 1, q = p ^ 1;

    v16h ah[4];
#pragma unroll
    for (int k = 0; k < 4; ++k) ah[k] = load_a_frag(lds_h[p], HID, lane, k * 32);
    v16h ax = load_a_frag(lds_x[p], DIN, lane, 0);

    v8f ar  = {0, 0, 0, 0, 0, 0, 0, 0};
    v8f az  = ar, axn = ar, ahn = ar;
    ar  = wmma_f16(ax, bxr, ar);
    az  = wmma_f16(ax, bxz, az);
    axn = wmma_f16(ax, bxn, axn);
#pragma unroll
    for (int k = 0; k < 4; ++k) {
      ar  = wmma_f16(ah[k], bhr[k], ar);
      az  = wmma_f16(ah[k], bhz[k], az);
      ahn = wmma_f16(ah[k], bhn[k], ahn);
    }

    if (s + 1 < SEQ) {                    // stage x(s+1) into the other buffer
      const int m = t >> 4, c = t & 15;
      const float2 xv =
          *(const float2*)(x + ((size_t)(b0 + m) * SEQ + (s + 1)) * DIN + 2 * c);
      lds_x[q][m * DIN + 2 * c]     = (half_t)xv.x;
      lds_x[q][m * DIN + 2 * c + 1] = (half_t)xv.y;
    }

    const size_t gb = ((size_t)s * BATCH + b0) * HID;
#pragma unroll
    for (int v = 0; v < 8; ++v) {
      float r  = fast_sigmoid(ar[v] + br);
      float z  = fast_sigmoid(az[v] + bz);
      float n  = fast_tanh((axn[v] + bxnb) + r * (ahn[v] + bhnb));
      float hh = (1.0f - z) * n + z * hprev[v];
      hprev[v] = hh;
      half_t hv = (half_t)hh;
      lds_h[q][(v + mb) * HID + j] = hv;            // next-step A matrix
      h1out[gb + (size_t)(v + mb) * HID + j] = hv;  // layer-1 input
    }
  }
}

// ---------------------------------------------------------------------------
// Layer 1: h1[S,B,128] f16 -> out[b*S+s] = relu(h2).W_lin + b_lin.
// W_hh fragments register-resident (critical recurrence path); W_ih stays in
// LDS (x-chains tolerate LDS latency). Double-buffered h/x/out tiles.
// ---------------------------------------------------------------------------
__global__ __launch_bounds__(TPB) void gru_layer1_kernel(
    const half_t* __restrict__ h1,
    const float* __restrict__ Wih,   // [384,128]
    const float* __restrict__ Whh,   // [384,128]
    const float* __restrict__ bih,
    const float* __restrict__ bhh,
    const float* __restrict__ Wlin,  // [1,128]
    const float* __restrict__ blin,  // [1]
    float* __restrict__ out) {
  extern __shared__ char smem[];
  half_t* ldsWih  = (half_t*)smem;                   // 384*128 f16
  half_t* lds_h0  = ldsWih + G3 * HID;               // [2][16*128]
  half_t* lds_h1  = lds_h0 + 16 * HID;
  half_t* lds_x0  = lds_h1 + 16 * HID;               // [2][16*128]
  half_t* lds_x1  = lds_x0 + 16 * HID;
  float*  lds_out = (float*)(lds_x1 + 16 * HID);     // [2][16]

  const int t    = threadIdx.x;
  const int lane = t & 31;
  const int w    = t >> 5;
  const int b0   = blockIdx.x * 16;

  for (int i = t; i < G3 * HID; i += TPB) ldsWih[i] = (half_t)Wih[i];

  v16h bhr[4], bhz[4], bhn[4];                        // 96 VGPRs, loop-invariant
#pragma unroll
  for (int k = 0; k < 4; ++k) {
    bhr[k] = load_b_frag_gmem(Whh, HID, lane, w,      k * 32);
    bhz[k] = load_b_frag_gmem(Whh, HID, lane, 8 + w,  k * 32);
    bhn[k] = load_b_frag_gmem(Whh, HID, lane, 16 + w, k * 32);
  }

  {
    uint4 z; z.x = z.y = z.z = z.w = 0u;
    ((uint4*)lds_h0)[t] = z;
  }
  if (t < 32) lds_out[t] = 0.0f;
  ((uint4*)lds_x0)[t] = ((const uint4*)(h1 + (size_t)b0 * HID))[t];

  const int   j    = (w << 4) + (lane & 15);
  const float br   = bih[j] + bhh[j];
  const float bz   = bih[HID + j] + bhh[HID + j];
  const float bxnb = bih[2 * HID + j];
  const float bhnb = bhh[2 * HID + j];
  const float wl   = Wlin[j];
  const float bl   = blin[0];
  const int   mb   = (lane >> 4) << 3;

  float hprev[8];
#pragma unroll
  for (int v = 0; v < 8; ++v) hprev[v] = 0.0f;

  for (int s = 0; s < SEQ; ++s) {
    __syncthreads();                       // step-(s-1) writes/adds visible
    const int p = s & 1, q = p ^ 1;
    half_t* hbuf_r = p ? lds_h1 : lds_h0;
    half_t* hbuf_w = p ? lds_h0 : lds_h1;
    half_t* xbuf_r = p ? lds_x1 : lds_x0;
    half_t* xbuf_w = p ? lds_x0 : lds_x1;

    if (s > 0 && t < 16) {                 // adds for step s-1 went to buf q
      out[(size_t)(b0 + t) * SEQ + (s - 1)] = lds_out[q * 16 + t] + bl;
      lds_out[q * 16 + t] = 0.0f;          // reused at step s+1, after barrier
    }

    v16h ah[4], axk[4];
#pragma unroll
    for (int k = 0; k < 4; ++k) {
      ah[k]  = load_a_frag(hbuf_r, HID, lane, k * 32);
      axk[k] = load_a_frag(xbuf_r, HID, lane, k * 32);
    }

    v8f ar  = {0, 0, 0, 0, 0, 0, 0, 0};
    v8f az  = ar, axn = ar, ahn = ar;
#pragma unroll
    for (int k = 0; k < 4; ++k) {
      ar  = wmma_f16(ah[k], bhr[k], ar);                 // critical recurrence
      az  = wmma_f16(ah[k], bhz[k], az);
      ahn = wmma_f16(ah[k], bhn[k], ahn);
      ar  = wmma_f16(axk[k], load_b_frag(ldsWih, HID, lane, w,      k * 32), ar);
      az  = wmma_f16(axk[k], load_b_frag(ldsWih, HID, lane, 8 + w,  k * 32), az);
      axn = wmma_f16(axk[k], load_b_frag(ldsWih, HID, lane, 16 + w, k * 32), axn);
    }

    if (s + 1 < SEQ) {                     // stage next h1 step (4KB burst)
      ((uint4*)xbuf_w)[t] =
          ((const uint4*)(h1 + ((size_t)(s + 1) * BATCH + b0) * HID))[t];
    }

    float part[8];
#pragma unroll
    for (int v = 0; v < 8; ++v) {
      float r  = fast_sigmoid(ar[v] + br);
      float z  = fast_sigmoid(az[v] + bz);
      float n  = fast_tanh((axn[v] + bxnb) + r * (ahn[v] + bhnb));
      float hh = (1.0f - z) * n + z * hprev[v];
      hprev[v] = hh;
      hbuf_w[(v + mb) * HID + j] = (half_t)hh;
      part[v] = fmaxf(hh, 0.0f) * wl;      // fused relu + linear partial
    }
#pragma unroll
    for (int mask = 1; mask <= 8; mask <<= 1) {
#pragma unroll
      for (int v = 0; v < 8; ++v) part[v] += __shfl_xor(part[v], mask, 32);
    }
    if ((lane & 15) == 0) {
#pragma unroll
      for (int v = 0; v < 8; ++v) atomicAdd(&lds_out[p * 16 + v + mb], part[v]);
    }
  }

  __syncthreads();
  if (t < 16)                              // last step's adds are in buf (SEQ-1)&1
    out[(size_t)(b0 + t) * SEQ + (SEQ - 1)] = lds_out[((SEQ - 1) & 1) * 16 + t] + bl;
}

// ---------------------------------------------------------------------------
extern "C" void kernel_launch(void* const* d_in, const int* in_sizes, int n_in,
                              void* d_out, int out_size, void* d_ws,
                              size_t ws_size, hipStream_t stream) {
  (void)in_sizes; (void)n_in; (void)out_size; (void)ws_size;
  const float* x    = (const float*)d_in[0];
  const float* Wih0 = (const float*)d_in[1];
  const float* Whh0 = (const float*)d_in[2];
  const float* bih0 = (const float*)d_in[3];
  const float* bhh0 = (const float*)d_in[4];
  const float* Wih1 = (const float*)d_in[5];
  const float* Whh1 = (const float*)d_in[6];
  const float* bih1 = (const float*)d_in[7];
  const float* bhh1 = (const float*)d_in[8];
  const float* Wlin = (const float*)d_in[9];
  const float* blin = (const float*)d_in[10];
  float*  out = (float*)d_out;
  half_t* h1  = (half_t*)d_ws;  // [SEQ][BATCH][HID] f16 = 78.6 MB

  const int SMEM1 =
      (int)((G3 * HID + 4 * 16 * HID) * sizeof(half_t) + 32 * sizeof(float));

  hipFuncSetAttribute((const void*)gru_layer1_kernel,
                      hipFuncAttributeMaxDynamicSharedMemorySize, SMEM1);

  gru_layer0_kernel<<<dim3(BATCH / 16), dim3(TPB), 0, stream>>>(
      x, Wih0, Whh0, bih0, bhh0, h1);
  gru_layer1_kernel<<<dim3(BATCH / 16), dim3(TPB), SMEM1, stream>>>(
      h1, Wih1, Whh1, bih1, bhh1, Wlin, blin, out);
}
